// GatedGCNConv_21887153340605
// MI455X (gfx1250) — matrix operand order repaired and checked by reading, not compiled, not run on hardware
//
#include <hip/hip_runtime.h>
#include <hip/hip_bf16.h>

// ---------------------------------------------------------------------------
// GatedGCNConv for MI455X (gfx1250, wave32)
//   support = x @ W ; gate = x @ Wg                (WMMA bf16x3 emulated fp32)
//   agg_*   = segment_sum(edge_vals * {support,gate}[cols], rows)  (L2 atomics)
//   out     = sigmoid(agg_gate) * agg_support
// ---------------------------------------------------------------------------

#define N_NODES 50000
#define N_EDGES 800000
#define IN_CH   256
#define OUT_CH  128
#define KTILES  (IN_CH / 32)   // 8 K-steps of 32 (bf16 WMMA K)
#define NTILES  (OUT_CH / 16)  // 8 N-tiles of 16

typedef __attribute__((ext_vector_type(16))) __bf16        v16bf;
typedef __attribute__((ext_vector_type(16))) unsigned short v16us;
typedef __attribute__((ext_vector_type(8)))  float          v8f;

static __device__ __forceinline__ unsigned short f32_to_bf16_rne(float f) {
    unsigned u = __builtin_bit_cast(unsigned, f);
    unsigned r = u + 0x7fffu + ((u >> 16) & 1u);   // round-to-nearest-even
    return (unsigned short)(r >> 16);
}
static __device__ __forceinline__ float bf16_to_f32(unsigned short h) {
    return __builtin_bit_cast(float, (unsigned)h << 16);
}

// ---------------------------------------------------------------------------
// Kernel 0: pack W (mat=0) and Wg (mat=1) [256x128 row-major fp32] into the
// WMMA 16-bit B-operand lane layout (ISA 7.12.2), split into bf16 hi/lo.
// Flat ushort index = (((mat*8 + ktile)*8 + ntile)*32 + lane)*16 + j
//   element j = W[(ktile*32 + ((j&8)?16:0) + ((lane&16)?8:0) + (j&7))*128 + ntile*16 + (lane&15)]
// ---------------------------------------------------------------------------
__global__ void pack_weights_kernel(const float* __restrict__ w,
                                    const float* __restrict__ gw,
                                    unsigned short* __restrict__ bhi,
                                    unsigned short* __restrict__ blo) {
    int idx   = blockIdx.x * blockDim.x + threadIdx.x;   // 0..65535
    int j     = idx & 15;
    int lane  = (idx >> 4) & 31;
    int ntile = (idx >> 9) & 7;
    int ktile = (idx >> 12) & 7;
    int mat   = (idx >> 15) & 1;
    int n     = ntile * 16 + (lane & 15);
    int krow  = ktile * 32 + ((j & 8) ? 16 : 0) + ((lane & 16) ? 8 : 0) + (j & 7);
    const float* src = mat ? gw : w;
    float f = src[krow * OUT_CH + n];
    unsigned short h = f32_to_bf16_rne(f);
    unsigned short l = f32_to_bf16_rne(f - bf16_to_f32(h));
    bhi[idx] = h;
    blo[idx] = l;
}

// ---------------------------------------------------------------------------
// Kernel 1: zero both segment-sum accumulators (agg_support lives in d_out).
// ---------------------------------------------------------------------------
__global__ void zero2_kernel(float* __restrict__ a, float* __restrict__ b, int n) {
    int i = blockIdx.x * blockDim.x + threadIdx.x;
    if (i < n) { a[i] = 0.0f; b[i] = 0.0f; }
}

// ---------------------------------------------------------------------------
// Kernel 2: fused dual GEMM. 128 threads = 4 waves; wave w owns rows
// [blockIdx*64 + w*16, +16) x all 128 cols. fp32 emulated via bf16 hi/lo:
//   acc += Ahi*Bhi + Alo*Bhi + Ahi*Blo   (f32 accumulate)
// ---------------------------------------------------------------------------
__global__ void __launch_bounds__(128, 1)
gemm_dual_kernel(const float* __restrict__ x,
                 const unsigned short* __restrict__ bhi,
                 const unsigned short* __restrict__ blo,
                 float* __restrict__ support,
                 float* __restrict__ gate) {
    const int lane    = threadIdx.x & 31;
    const int wave    = threadIdx.x >> 5;
    const int rowBase = blockIdx.x * 64 + wave * 16;
    const int m       = lane & 15;
    int arow = rowBase + m;
    if (arow >= N_NODES) arow = N_NODES - 1;           // clamp (stores guarded)
    const int khalf = (lane & 16) ? 8 : 0;             // A layout: upper lanes K+8
    const float* aptr = x + (size_t)arow * IN_CH + khalf;

    v8f accS[NTILES], accG[NTILES];
#pragma unroll
    for (int t = 0; t < NTILES; t++) { accS[t] = (v8f)0.0f; accG[t] = (v8f)0.0f; }

    const v16us* bhiV = (const v16us*)bhi;
    const v16us* bloV = (const v16us*)blo;

    for (int ks = 0; ks < KTILES; ks++) {
        // A fragment: lanes 0-15 hold K = {0..7,16..23}, lanes 16-31 K = {8..15,24..31}
        const float4* ap = (const float4*)(aptr + ks * 32);
        float4 f0 = ap[0], f1 = ap[1];      // K khalf+0..7
        float4 f2 = ap[4], f3 = ap[5];      // K khalf+16..23
        float av[16] = { f0.x, f0.y, f0.z, f0.w,  f1.x, f1.y, f1.z, f1.w,
                         f2.x, f2.y, f2.z, f2.w,  f3.x, f3.y, f3.z, f3.w };
        v16us ahu, alu;
#pragma unroll
        for (int i = 0; i < 16; i++) {
            unsigned short h = f32_to_bf16_rne(av[i]);
            ahu[i] = h;
            alu[i] = f32_to_bf16_rne(av[i] - bf16_to_f32(h));
        }
        v16bf ahi = __builtin_bit_cast(v16bf, ahu);
        v16bf alo = __builtin_bit_cast(v16bf, alu);

#pragma unroll
        for (int nt = 0; nt < NTILES; nt++) {
            const int biS = ((0 * KTILES + ks) * NTILES + nt) * 32 + lane;
            const int biG = ((1 * KTILES + ks) * NTILES + nt) * 32 + lane;
            v16bf bhS = __builtin_bit_cast(v16bf, bhiV[biS]);
            v16bf blS = __builtin_bit_cast(v16bf, bloV[biS]);
            v16bf bhG = __builtin_bit_cast(v16bf, bhiV[biG]);
            v16bf blG = __builtin_bit_cast(v16bf, bloV[biG]);
            accS[nt] = __builtin_amdgcn_wmma_f32_16x16x32_bf16(
                false, ahi, false, bhS, (short)0, accS[nt], false, false);
            accS[nt] = __builtin_amdgcn_wmma_f32_16x16x32_bf16(
                false, alo, false, bhS, (short)0, accS[nt], false, false);
            accS[nt] = __builtin_amdgcn_wmma_f32_16x16x32_bf16(
                false, ahi, false, blS, (short)0, accS[nt], false, false);
            accG[nt] = __builtin_amdgcn_wmma_f32_16x16x32_bf16(
                false, ahi, false, bhG, (short)0, accG[nt], false, false);
            accG[nt] = __builtin_amdgcn_wmma_f32_16x16x32_bf16(
                false, alo, false, bhG, (short)0, accG[nt], false, false);
            accG[nt] = __builtin_amdgcn_wmma_f32_16x16x32_bf16(
                false, ahi, false, blG, (short)0, accG[nt], false, false);
        }
    }

    // C/D layout: VGPR r -> row rowBase + r (+8 for lanes 16-31), col = nt*16 + (lane&15)
    const int rOff = (lane & 16) ? 8 : 0;
#pragma unroll
    for (int nt = 0; nt < NTILES; nt++) {
        int col = nt * 16 + m;
#pragma unroll
        for (int r = 0; r < 8; r++) {
            int row = rowBase + rOff + r;
            if (row < N_NODES) {
                support[(size_t)row * OUT_CH + col] = accS[nt][r];
                gate[(size_t)row * OUT_CH + col]    = accG[nt][r];
            }
        }
    }
}

// ---------------------------------------------------------------------------
// Kernel 3: edge-parallel SpMM for both matrices. One wave per edge; 32 lanes
// x float4 = 128 channels. Gathers hit L2 (support+gate = 51 MB < 192 MB L2);
// scatter via no-return global_atomic_add_f32.
// ---------------------------------------------------------------------------
__global__ void __launch_bounds__(256)
spmm_edges_kernel(const int* __restrict__ erows,
                  const int* __restrict__ ecols,
                  const float* __restrict__ evals,
                  const float* __restrict__ support,
                  const float* __restrict__ gate,
                  float* __restrict__ aggS,
                  float* __restrict__ aggG) {
    int wave = threadIdx.x >> 5;
    int lane = threadIdx.x & 31;
    int e = blockIdx.x * 8 + wave;
    if (e >= N_EDGES) return;
    int   r = erows[e];
    int   c = ecols[e];
    float v = evals[e];
    int   co = lane * 4;
    float4 s = *(const float4*)(support + (size_t)c * OUT_CH + co);
    float4 g = *(const float4*)(gate    + (size_t)c * OUT_CH + co);
    float* ps = aggS + (size_t)r * OUT_CH + co;
    float* pg = aggG + (size_t)r * OUT_CH + co;
    unsafeAtomicAdd(ps + 0, v * s.x);
    unsafeAtomicAdd(ps + 1, v * s.y);
    unsafeAtomicAdd(ps + 2, v * s.z);
    unsafeAtomicAdd(ps + 3, v * s.w);
    unsafeAtomicAdd(pg + 0, v * g.x);
    unsafeAtomicAdd(pg + 1, v * g.y);
    unsafeAtomicAdd(pg + 2, v * g.z);
    unsafeAtomicAdd(pg + 3, v * g.w);
}

// ---------------------------------------------------------------------------
// Kernel 4: out = sigmoid(agg_gate) * agg_support   (agg_support in d_out)
// ---------------------------------------------------------------------------
__global__ void finalize_kernel(float* __restrict__ out,
                                const float* __restrict__ aggG, int n4) {
    int i = blockIdx.x * blockDim.x + threadIdx.x;
    if (i < n4) {
        float4 s = ((const float4*)out)[i];
        float4 g = ((const float4*)aggG)[i];
        float4 o;
        o.x = s.x / (1.0f + expf(-g.x));
        o.y = s.y / (1.0f + expf(-g.y));
        o.z = s.z / (1.0f + expf(-g.z));
        o.w = s.w / (1.0f + expf(-g.w));
        ((float4*)out)[i] = o;
    }
}

// ---------------------------------------------------------------------------
extern "C" void kernel_launch(void* const* d_in, const int* in_sizes, int n_in,
                              void* d_out, int out_size, void* d_ws, size_t ws_size,
                              hipStream_t stream) {
    const float* x     = (const float*)d_in[0];
    const int*   erows = (const int*)d_in[1];
    const int*   ecols = (const int*)d_in[2];
    const float* evals = (const float*)d_in[3];
    const float* w     = (const float*)d_in[4];
    const float* gw    = (const float*)d_in[5];
    float* out = (float*)d_out;

    const size_t NC = (size_t)N_NODES * OUT_CH;   // 6.4M floats
    float* support = (float*)d_ws;                // 25.6 MB
    float* gate    = support + NC;                // 25.6 MB
    float* aggG    = gate + NC;                   // 25.6 MB
    unsigned short* bhi = (unsigned short*)(aggG + NC);  // 128 KB
    unsigned short* blo = bhi + 65536;                   // 128 KB
    // agg_support accumulates directly in d_out.

    // 0) pack weights into WMMA B lane layout (bf16 hi/lo)
    pack_weights_kernel<<<65536 / 256, 256, 0, stream>>>(w, gw, bhi, blo);
    // 1) zero accumulators
    zero2_kernel<<<(int)((NC + 255) / 256), 256, 0, stream>>>(out, aggG, (int)NC);
    // 2) dual GEMM (4 waves/block, 64 rows/block)
    gemm_dual_kernel<<<(N_NODES + 63) / 64, 128, 0, stream>>>(x, bhi, blo, support, gate);
    // 3) edge scatter-add (8 waves/block = 8 edges/block)
    spmm_edges_kernel<<<(N_EDGES + 7) / 8, 256, 0, stream>>>(
        erows, ecols, evals, support, gate, out, aggG);
    // 4) gating
    finalize_kernel<<<(int)((NC / 4 + 255) / 256), 256, 0, stream>>>(out, aggG, (int)(NC / 4));
}
